// MultiHeadLatentAttention_76836964925534
// MI455X (gfx1250) — matrix operand order repaired and checked by reading, not compiled, not run on hardware
//
#include <hip/hip_runtime.h>
#include <hip/hip_bf16.h>

typedef __attribute__((ext_vector_type(16))) __bf16 v16bf;
typedef __attribute__((ext_vector_type(8)))  float  v8f;
typedef __attribute__((ext_vector_type(4)))  unsigned int u32x4;
typedef __attribute__((ext_vector_type(8)))  int i32x8;
typedef __attribute__((ext_vector_type(4)))  int i32x4;
typedef unsigned short u16;
typedef unsigned int   u32;

#define B_    2
#define S_    2048
#define HID_  2048
#define NH_   16
#define DN_   128
#define DR_   64
#define DQK_  192
#define DCKV_ 512
#define DCQ_  1536
#define DV_   128
#define MROWS (B_*S_)                 // 4096
#define SCALE_ 0.07216878364870323f   // 192^-0.5

#if defined(__has_builtin)
# if __has_builtin(__builtin_amdgcn_tensor_load_to_lds) && __has_builtin(__builtin_amdgcn_s_wait_tensorcnt)
#  define HAVE_TDM 1
# endif
#endif
#ifndef HAVE_TDM
# define HAVE_TDM 0
#endif

__device__ __forceinline__ u16 f2bf(float f) {
  u32 u = __float_as_uint(f);
  u32 r = u + 0x7FFFu + ((u >> 16) & 1u);   // round-to-nearest-even
  return (u16)(r >> 16);
}

union FragAB { v16bf v; uint4 q[2]; };

__device__ __forceinline__ v8f zero8() {
  v8f z;
#pragma unroll
  for (int i = 0; i < 8; ++i) z[i] = 0.f;
  return z;
}

#if HAVE_TDM
// Issue a 2D TDM load global->LDS. Dims/strides in 2-byte elements.
// D# layout per CDNA5 ISA.
__device__ __forceinline__ void tdm_load_2d(
    u32 lds_byte, const u16* gptr, u32 tensor_d0, u32 tensor_d1,
    u32 tile_d0, u32 tile_d1, unsigned long long stride0_elems)
{
  unsigned long long ga = (unsigned long long)(uintptr_t)gptr;
  u32x4 g0;
  g0[0] = 1u;                                              // count=1, user desc
  g0[1] = lds_byte;                                        // lds_addr
  g0[2] = (u32)ga;                                         // global_addr[31:0]
  g0[3] = (u32)((ga >> 32) & 0x01FFFFFFu) | (2u << 30);    // gaddr[56:32] | type=2
  i32x8 g1;
  g1[0] = (int)(1u << 16);                                 // data_size = 2 bytes
  g1[1] = (int)((tensor_d0 & 0xFFFFu) << 16);              // tensor_dim0[15:0]
  g1[2] = (int)((tensor_d0 >> 16) | ((tensor_d1 & 0xFFFFu) << 16));
  g1[3] = (int)((tensor_d1 >> 16) | (tile_d0 << 16));      // tile_dim0
  g1[4] = (int)(tile_d1 & 0xFFFFu);                        // tile_dim1, tile_dim2=0
  g1[5] = (int)(u32)(stride0_elems & 0xFFFFFFFFull);       // dim0_stride[31:0]
  g1[6] = (int)(u32)((stride0_elems >> 32) & 0xFFFFull);   // dim0_stride[47:32]
  g1[7] = 0;
  i32x4 z4; z4[0] = 0; z4[1] = 0; z4[2] = 0; z4[3] = 0;
  i32x8 z8;
#pragma unroll
  for (int i = 0; i < 8; ++i) z8[i] = 0;
  // 6-arg form (clang-23 / therock headers): (g0, g1, g2, g3, g4, cpol)
  __builtin_amdgcn_tensor_load_to_lds(g0, g1, z4, z4, z8, 0);
}
#endif

// ---------------------------------------------------------------- fp32 -> bf16
__global__ void k_cvt_bf16(const float* __restrict__ in, u16* __restrict__ out, long n) {
  long i = (long)blockIdx.x * blockDim.x + threadIdx.x;
  long stride = (long)gridDim.x * blockDim.x;
  for (; i < n; i += stride) out[i] = f2bf(in[i]);
}

// -------------------------- fp32 [K][N] -> bf16 transposed [Npad][K], zero pad
__global__ __launch_bounds__(256) void k_transpose_bf16(
    const float* __restrict__ in, u16* __restrict__ out, int K, int N, int Npad)
{
  __shared__ float tile[32][33];
  (void)Npad;
  const int k0 = blockIdx.x * 32, n0 = blockIdx.y * 32;
  for (int r = threadIdx.y; r < 32; r += 8) {
    const int k = k0 + r, n = n0 + threadIdx.x;
    tile[r][threadIdx.x] = (n < N) ? in[(size_t)k * N + n] : 0.f;
  }
  __syncthreads();
  for (int r = threadIdx.y; r < 32; r += 8) {
    const int n = n0 + r, k = k0 + threadIdx.x;
    out[(size_t)n * K + k] = f2bf(tile[threadIdx.x][r]);
  }
}

// -------- V fp32 [B,S,NH,DV] -> bf16 Vt [B*NH][DV][S] (key-contiguous per dim)
__global__ __launch_bounds__(256) void k_transpose_v_bf16(
    const float* __restrict__ v, u16* __restrict__ vt)
{
  __shared__ float tile[32][33];
  const int bh = blockIdx.z, b = bh / NH_, h = bh % NH_;
  const int s0 = blockIdx.x * 32, n0 = blockIdx.y * 32;
  for (int r = threadIdx.y; r < 32; r += 8)
    tile[r][threadIdx.x] =
        v[((size_t)(b * S_ + s0 + r) * NH_ + h) * DV_ + n0 + threadIdx.x];
  __syncthreads();
  for (int r = threadIdx.y; r < 32; r += 8)
    vt[((size_t)bh * DV_ + n0 + r) * S_ + s0 + threadIdx.x] = f2bf(tile[threadIdx.x][r]);
}

// ------------------------------------------------------- generic bf16 WMMA GEMM
// C[M,N] fp32 = A[M,K] bf16 (row major) @ Bt[N,K] bf16 (pre-transposed weights)
// M % 128 == 0, N % 128 == 0, K % 32 == 0. Double-buffered LDS pipeline.
__global__ __launch_bounds__(256) void k_gemm_bf16(
    const u16* __restrict__ A, const u16* __restrict__ Bt, float* __restrict__ C,
    int N, int K)
{
  __shared__ __align__(16) u16 lsA[2][128 * 32];   // [m][k]
  __shared__ __align__(16) u16 lsB[2][128 * 32];   // [n][k]
  const int tid  = threadIdx.x;
  const int wave = tid >> 5, lane = tid & 31;
  const int half = lane >> 4, l16 = lane & 15;
  const int m0 = blockIdx.x * 128, n0 = blockIdx.y * 128;
  const int row = tid >> 1, kk = (tid & 1) * 16;   // staging coords (32B/thread)
  const u16* gA = A  + (size_t)(m0 + row) * K + kk;
  const u16* gB = Bt + (size_t)(n0 + row) * K + kk;

  v8f acc[8];
#pragma unroll
  for (int f = 0; f < 8; ++f) acc[f] = zero8();

  { // prologue: stage tile 0
    *(uint4*)(lsA[0] + row * 32 + kk)     = *(const uint4*)(gA);
    *(uint4*)(lsA[0] + row * 32 + kk + 8) = *(const uint4*)(gA + 8);
    *(uint4*)(lsB[0] + row * 32 + kk)     = *(const uint4*)(gB);
    *(uint4*)(lsB[0] + row * 32 + kk + 8) = *(const uint4*)(gB + 8);
  }
  __syncthreads();

  const int T = K >> 5;
  for (int t = 0; t < T; ++t) {
    uint4 a0, a1, b0, b1;
    if (t + 1 < T) {                       // prefetch next tile into registers
      const u16* pA = gA + (size_t)(t + 1) * 32;
      const u16* pB = gB + (size_t)(t + 1) * 32;
      a0 = *(const uint4*)(pA); a1 = *(const uint4*)(pA + 8);
      b0 = *(const uint4*)(pB); b1 = *(const uint4*)(pB + 8);
    }
    const u16* la = lsA[t & 1];
    const u16* lb = lsB[t & 1];

    FragAB a;   // A 16x32 fragment (contiguous K: 0-15 vs 16-31 by half)
    {
      const u16* ap = la + (wave * 16 + l16) * 32;
      a.q[0] = *(const uint4*)(ap + half * 16);
      a.q[1] = *(const uint4*)(ap + half * 16 + 8);
    }
    FragAB bfr[8];  // preload all B fragments
#pragma unroll
    for (int f = 0; f < 8; ++f) {
      const u16* bp = lb + (f * 16 + l16) * 32 + half * 16;
      bfr[f].q[0] = *(const uint4*)(bp);
      bfr[f].q[1] = *(const uint4*)(bp + 8);
    }
    // scheduling fence: keep all ds_loads before the WMMA burst -> one dscnt wait
    asm volatile("" ::: "memory");
#pragma unroll
    for (int f = 0; f < 8; ++f)
      acc[f] = __builtin_amdgcn_wmma_f32_16x16x32_bf16(
          false, a.v, false, bfr[f].v, (short)0, acc[f], false, false);

    if (t + 1 < T) {                       // drain prefetch into the other buffer
      u16* da = lsA[(t + 1) & 1] + row * 32 + kk;
      u16* db = lsB[(t + 1) & 1] + row * 32 + kk;
      *(uint4*)(da) = a0; *(uint4*)(da + 8) = a1;
      *(uint4*)(db) = b0; *(uint4*)(db + 8) = b1;
    }
    __syncthreads();
  }

#pragma unroll
  for (int f = 0; f < 8; ++f) {
#pragma unroll
    for (int v = 0; v < 8; ++v) {
      const int r = m0 + wave * 16 + v + half * 8;
      const int c = n0 + f * 16 + l16;
      C[(size_t)r * N + c] = acc[f][v];
    }
  }
}

// ---------------------------------------------------------------- RMSNorm (row)
__global__ __launch_bounds__(256) void k_rmsnorm_bf16(
    const float* __restrict__ X, const float* __restrict__ w, u16* __restrict__ out, int cols)
{
  __shared__ float red[256];
  const int row = blockIdx.x, tid = threadIdx.x;
  const float* xr = X + (size_t)row * cols;
  float ss = 0.f;
  for (int c = tid; c < cols; c += 256) { float v = xr[c]; ss += v * v; }
  red[tid] = ss; __syncthreads();
  for (int s = 128; s > 0; s >>= 1) { if (tid < s) red[tid] += red[tid + s]; __syncthreads(); }
  const float rstd = rsqrtf(red[0] / (float)cols + 1e-6f);
  u16* orow = out + (size_t)row * cols;
  for (int c = tid; c < cols; c += 256) orow[c] = f2bf(xr[c] * rstd * w[c]);
}

// ------------------------------------------ assemble q = [q_nope, rope(q_rope)]
__global__ void k_assemble_q(const float* __restrict__ qc, const float* __restrict__ qr,
                             const float* __restrict__ cosT, const float* __restrict__ sinT,
                             u16* __restrict__ qbf)
{
  const int bs = blockIdx.x, h = blockIdx.y, d = threadIdx.x;   // d in [0,64)
  const int b = bs / S_, s = bs % S_;
  const float* qrp = qr + (size_t)bs * (NH_ * DR_) + h * DR_;
  float x  = qrp[d];
  float xo = (d < 32) ? -qrp[d + 32] : qrp[d - 32];
  float rope = x * cosT[s * DR_ + d] + xo * sinT[s * DR_ + d];
  u16* dst = qbf + ((size_t)(b * NH_ + h) * S_ + s) * DQK_;
  const float* qcp = qc + (size_t)bs * (NH_ * DN_) + h * DN_;
  dst[d]        = f2bf(qcp[d]);
  dst[d + 64]   = f2bf(qcp[d + 64]);
  dst[DN_ + d]  = f2bf(rope);
}

// -------------------------------- assemble k = [k_nope, broadcast rope(k_rope)]
__global__ void k_assemble_k(const float* __restrict__ kc, const float* __restrict__ kr,
                             const float* __restrict__ cosT, const float* __restrict__ sinT,
                             u16* __restrict__ kbf)
{
  const int bs = blockIdx.x, h = blockIdx.y, d = threadIdx.x;
  const int b = bs / S_, s = bs % S_;
  const float* krp = kr + (size_t)bs * 128;      // padded stride
  float x  = krp[d];
  float xo = (d < 32) ? -krp[d + 32] : krp[d - 32];
  float rope = x * cosT[s * DR_ + d] + xo * sinT[s * DR_ + d];
  u16* dst = kbf + ((size_t)(b * NH_ + h) * S_ + s) * DQK_;
  const float* kcp = kc + (size_t)bs * (NH_ * DN_) + h * DN_;
  dst[d]        = f2bf(kcp[d]);
  dst[d + 64]   = f2bf(kcp[d + 64]);
  dst[DN_ + d]  = f2bf(rope);
}

// ------------------------------------------------- causal flash attention (WMMA)
// Q,K: [B*NH, S, DQK] bf16; Vt: [B*NH, DV, S] bf16; O: [B, S, NH, DV] fp32
__global__ __launch_bounds__(256) void k_flash(
    const u16* __restrict__ Q, const u16* __restrict__ Kc, const u16* __restrict__ Vt,
    float* __restrict__ O)
{
  __shared__ __align__(16) u16 lsK[32 * DQK_];    // [key][dqk]  (== B-frag layout)
  __shared__ __align__(16) u16 lsV[DV_ * 32];     // [vdim][key]
  __shared__ __align__(16) u16 lsP[8][16 * 32];   // per-wave P tile

  const int tid = threadIdx.x, wave = tid >> 5, lane = tid & 31;
  const int half = lane >> 4, l16 = lane & 15;
  const int bh = blockIdx.y, b = bh / NH_, h = bh % NH_;
  const int qbase = blockIdx.x * 128 + wave * 16;
  const u16* Qh = Q  + (size_t)bh * S_ * DQK_;
  const u16* Kh = Kc + (size_t)bh * S_ * DQK_;
  const u16* Vh = Vt + (size_t)bh * DV_ * S_;

#if HAVE_TDM
  // LDS aperture: low 32 bits of the flat address are the LDS byte address
  const u32 lsK_byte = (u32)(uintptr_t)(const void*)lsK;
  const u32 lsV_byte = (u32)(uintptr_t)(const void*)lsV;
#else
  const int srow = tid >> 1, skk = (tid & 1) * 16;   // staging coords
#endif

  FragAB qf[6];                                   // Q rows once, whole DQK=192
#pragma unroll
  for (int c = 0; c < 6; ++c) {
    const u16* qp = Qh + (size_t)(qbase + l16) * DQK_ + c * 32;
    qf[c].q[0] = *(const uint4*)(qp + half * 16);
    qf[c].q[1] = *(const uint4*)(qp + half * 16 + 8);
  }

  v8f acc[8];
#pragma unroll
  for (int f = 0; f < 8; ++f) acc[f] = zero8();
  float mrow[8], lrow[8];
#pragma unroll
  for (int v = 0; v < 8; ++v) { mrow[v] = -3.0e38f; lrow[v] = 0.f; }

  const int nkt = blockIdx.x * 4 + 4;             // causal: keys <= block max row
  for (int kt = 0; kt < nkt; ++kt) {
    const int kbase = kt * 32;
#if HAVE_TDM
    if (wave == 0) {
      // K tile: 32 rows x 192 elems, row stride 192 (contiguous 12KB in LDS)
      tdm_load_2d(lsK_byte, Kh + (size_t)kbase * DQK_, DQK_, 32, DQK_, 32, DQK_);
      // V tile: 128 rows x 32 keys, row stride S (8KB in LDS, [vdim][key])
      tdm_load_2d(lsV_byte, Vh + kbase, 32, DV_, 32, DV_, S_);
      __builtin_amdgcn_s_wait_tensorcnt(0);
    }
#else
    for (int i = tid; i < 32 * (DQK_ / 8); i += 256) {       // stage K (16B copies)
      const int key = i / (DQK_ / 8), c8 = (i % (DQK_ / 8)) * 8;
      *(uint4*)(lsK + key * DQK_ + c8) =
          *(const uint4*)(Kh + (size_t)(kbase + key) * DQK_ + c8);
    }
    {                                                        // stage V (16B copies)
      const u16* src = Vh + (size_t)srow * S_ + kbase + skk;
      uint4* dst = (uint4*)(lsV + srow * 32 + skk);
      dst[0] = ((const uint4*)src)[0];
      dst[1] = ((const uint4*)src)[1];
    }
#endif
    __syncthreads();

    v8f s0 = zero8(), s1 = zero8();   // two interleaved accumulation chains
#pragma unroll
    for (int c = 0; c < 6; ++c) {
      FragAB ka, kb;
      const u16* kp0 = lsK + l16 * DQK_ + c * 32 + half * 16;
      const u16* kp1 = lsK + (16 + l16) * DQK_ + c * 32 + half * 16;
      ka.q[0] = *(const uint4*)kp0; ka.q[1] = *(const uint4*)(kp0 + 8);
      kb.q[0] = *(const uint4*)kp1; kb.q[1] = *(const uint4*)(kp1 + 8);
      s0 = __builtin_amdgcn_wmma_f32_16x16x32_bf16(false, qf[c].v, false, ka.v, (short)0, s0, false, false);
      s1 = __builtin_amdgcn_wmma_f32_16x16x32_bf16(false, qf[c].v, false, kb.v, (short)0, s1, false, false);
    }

    u16* pw = &lsP[wave][0];
#pragma unroll
    for (int v = 0; v < 8; ++v) {                 // online softmax per row
      const int qrow = qbase + v + half * 8;
      float a  = s0[v] * SCALE_;
      float bb = s1[v] * SCALE_;
      if (kbase + l16      > qrow) a  = -3.0e38f;
      if (kbase + 16 + l16 > qrow) bb = -3.0e38f;
      float x = fmaxf(a, bb);
      for (int m = 1; m < 16; m <<= 1) x = fmaxf(x, __shfl_xor(x, m, 32));
      const float mnew = fmaxf(mrow[v], x);
      const float corr = __expf(mrow[v] - mnew);
      const float p0 = __expf(a  - mnew);
      const float p1 = __expf(bb - mnew);
      float ps = p0 + p1;
      for (int m = 1; m < 16; m <<= 1) ps += __shfl_xor(ps, m, 32);
      lrow[v] = lrow[v] * corr + ps;
      mrow[v] = mnew;
#pragma unroll
      for (int f = 0; f < 8; ++f) acc[f][v] *= corr;
      const int prow = (v + half * 8) * 32;       // C-layout -> P tile row-major
      pw[prow + l16]      = f2bf(p0);
      pw[prow + 16 + l16] = f2bf(p1);
    }
    asm volatile("s_wait_dscnt 0" ::: "memory");  // wave-local LDS fence
    __builtin_amdgcn_wave_barrier();

    FragAB pf;                                    // reload P as A-fragment (16x32)
    {
      const u16* pp = pw + l16 * 32;
      pf.q[0] = *(const uint4*)(pp + half * 16);
      pf.q[1] = *(const uint4*)(pp + half * 16 + 8);
    }
#pragma unroll
    for (int g = 0; g < 2; ++g) {                 // O += P @ V, preload 4 frags/group
      FragAB vb[4];
#pragma unroll
      for (int f = 0; f < 4; ++f) {
        const u16* vp = lsV + ((g * 4 + f) * 16 + l16) * 32 + half * 16;
        vb[f].q[0] = *(const uint4*)vp; vb[f].q[1] = *(const uint4*)(vp + 8);
      }
      asm volatile("" ::: "memory");              // burst loads before WMMAs
#pragma unroll
      for (int f = 0; f < 4; ++f)
        acc[g * 4 + f] = __builtin_amdgcn_wmma_f32_16x16x32_bf16(
            false, pf.v, false, vb[f].v, (short)0, acc[g * 4 + f], false, false);
    }
    __syncthreads();
  }

#pragma unroll
  for (int v = 0; v < 8; ++v) {
    const float inv = lrow[v] > 0.f ? 1.f / lrow[v] : 0.f;
    const int r = qbase + v + half * 8;
    float* orow = O + ((size_t)(b * S_ + r) * NH_ + h) * DV_;
#pragma unroll
    for (int f = 0; f < 8; ++f) orow[f * 16 + l16] = acc[f][v] * inv;
  }
}

// =============================================================== host launcher
extern "C" void kernel_launch(void* const* d_in, const int* in_sizes, int n_in,
                              void* d_out, int out_size, void* d_ws, size_t ws_size,
                              hipStream_t stream)
{
  (void)in_sizes; (void)n_in; (void)out_size; (void)ws_size;
  const float* hidden  = (const float*)d_in[0];
  const float* cosT    = (const float*)d_in[1];
  const float* sinT    = (const float*)d_in[2];
  const float* Wq_down = (const float*)d_in[3];
  const float* q_norm  = (const float*)d_in[4];
  const float* Wqc_up  = (const float*)d_in[5];
  const float* Wqr_up  = (const float*)d_in[6];
  const float* Wkv_a   = (const float*)d_in[7];
  const float* Wk_rope = (const float*)d_in[8];
  const float* kv_norm = (const float*)d_in[9];
  const float* Wkc_up  = (const float*)d_in[10];
  const float* Wv_up   = (const float*)d_in[11];
  const float* Wo      = (const float*)d_in[12];

  char* p = (char*)d_ws;
  auto alloc = [&](size_t bytes) -> void* {
    void* r = (void*)p;
    p += (bytes + 255) & ~(size_t)255;
    return r;
  };

  u16*  hid_bf  = (u16*)alloc((size_t)MROWS * HID_ * 2);
  u16*  wqd_t   = (u16*)alloc((size_t)DCQ_ * HID_ * 2);
  u16*  wqc_t   = (u16*)alloc((size_t)(NH_ * DN_) * DCQ_ * 2);
  u16*  wqr_t   = (u16*)alloc((size_t)(NH_ * DR_) * DCQ_ * 2);
  u16*  wkva_t  = (u16*)alloc((size_t)DCKV_ * HID_ * 2);
  u16*  wkr_t   = (u16*)alloc((size_t)128 * HID_ * 2);          // padded 64->128
  u16*  wkc_t   = (u16*)alloc((size_t)(NH_ * DN_) * DCKV_ * 2);
  u16*  wv_t    = (u16*)alloc((size_t)(NH_ * DV_) * DCKV_ * 2);
  u16*  wo_t    = (u16*)alloc((size_t)HID_ * (NH_ * DV_) * 2);
  float* cq_f   = (float*)alloc((size_t)MROWS * DCQ_ * 4);
  u16*  cq_bf   = (u16*)alloc((size_t)MROWS * DCQ_ * 2);
  float* ckv_f  = (float*)alloc((size_t)MROWS * DCKV_ * 4);
  u16*  ckv_bf  = (u16*)alloc((size_t)MROWS * DCKV_ * 2);
  float* kr_f   = (float*)alloc((size_t)MROWS * 128 * 4);       // padded stride
  float* qc_f   = (float*)alloc((size_t)MROWS * NH_ * DN_ * 4);
  float* qr_f   = (float*)alloc((size_t)MROWS * NH_ * DR_ * 4);
  float* kc_f   = (float*)alloc((size_t)MROWS * NH_ * DN_ * 4);
  float* v_f    = (float*)alloc((size_t)MROWS * NH_ * DV_ * 4);
  u16*  q_bf    = (u16*)alloc((size_t)B_ * NH_ * S_ * DQK_ * 2);
  u16*  k_bf    = (u16*)alloc((size_t)B_ * NH_ * S_ * DQK_ * 2);
  u16*  vt_bf   = (u16*)alloc((size_t)B_ * NH_ * DV_ * S_ * 2);
  float* ao_f   = (float*)alloc((size_t)MROWS * NH_ * DV_ * 4);
  u16*  ao_bf   = (u16*)alloc((size_t)MROWS * NH_ * DV_ * 2);

  auto cvt = [&](const float* src, u16* dst, long n) {
    int blocks = (int)((n + 255) / 256); if (blocks > 4096) blocks = 4096;
    k_cvt_bf16<<<blocks, 256, 0, stream>>>(src, dst, n);
  };
  auto transw = [&](const float* W, u16* Wt, int K, int N, int Npad) {
    dim3 g(K / 32, Npad / 32);
    k_transpose_bf16<<<g, dim3(32, 8), 0, stream>>>(W, Wt, K, N, Npad);
  };
  auto gemm = [&](const u16* A, const u16* Bt, float* C, int M, int N, int K) {
    dim3 g(M / 128, N / 128);
    k_gemm_bf16<<<g, 256, 0, stream>>>(A, Bt, C, N, K);
  };

  // bf16 conversions: activations straight, weights transposed [Npad][K]
  cvt(hidden, hid_bf, (long)MROWS * HID_);
  transw(Wq_down, wqd_t,  HID_,  DCQ_,      DCQ_);
  transw(Wqc_up,  wqc_t,  DCQ_,  NH_ * DN_, NH_ * DN_);
  transw(Wqr_up,  wqr_t,  DCQ_,  NH_ * DR_, NH_ * DR_);
  transw(Wkv_a,   wkva_t, HID_,  DCKV_,     DCKV_);
  transw(Wk_rope, wkr_t,  HID_,  DR_,       128);
  transw(Wkc_up,  wkc_t,  DCKV_, NH_ * DN_, NH_ * DN_);
  transw(Wv_up,   wv_t,   DCKV_, NH_ * DV_, NH_ * DV_);
  transw(Wo,      wo_t,   NH_ * DV_, HID_,  HID_);

  // down projections + norms + rope key
  gemm(hid_bf, wqd_t,  cq_f,  MROWS, DCQ_,  HID_);
  k_rmsnorm_bf16<<<MROWS, 256, 0, stream>>>(cq_f, q_norm, cq_bf, DCQ_);
  gemm(hid_bf, wkva_t, ckv_f, MROWS, DCKV_, HID_);
  k_rmsnorm_bf16<<<MROWS, 256, 0, stream>>>(ckv_f, kv_norm, ckv_bf, DCKV_);
  gemm(hid_bf, wkr_t,  kr_f,  MROWS, 128,   HID_);

  // up projections
  gemm(cq_bf,  wqc_t, qc_f, MROWS, NH_ * DN_, DCQ_);
  gemm(cq_bf,  wqr_t, qr_f, MROWS, NH_ * DR_, DCQ_);
  gemm(ckv_bf, wkc_t, kc_f, MROWS, NH_ * DN_, DCKV_);
  gemm(ckv_bf, wv_t,  v_f,  MROWS, NH_ * DV_, DCKV_);

  // rope + head assembly + V transpose
  k_assemble_q<<<dim3(MROWS, NH_), 64, 0, stream>>>(qc_f, qr_f, cosT, sinT, q_bf);
  k_assemble_k<<<dim3(MROWS, NH_), 64, 0, stream>>>(kc_f, kr_f, cosT, sinT, k_bf);
  k_transpose_v_bf16<<<dim3(S_ / 32, DV_ / 32, B_ * NH_), dim3(32, 8), 0, stream>>>(v_f, vt_bf);

  // causal flash attention
  k_flash<<<dim3(S_ / 128, B_ * NH_), 256, 0, stream>>>(q_bf, k_bf, vt_bf, ao_f);

  // output projection
  cvt(ao_f, ao_bf, (long)MROWS * NH_ * DV_);
  gemm(ao_bf, wo_t, (float*)d_out, MROWS, HID_, NH_ * DV_);
}
